// LengthRegulator_4234837753892
// MI455X (gfx1250) — matrix operand order repaired and checked by reading, not compiled, not run on hardware
//
#include <hip/hip_runtime.h>

typedef __attribute__((ext_vector_type(16))) _Float16 v16h;
typedef __attribute__((ext_vector_type(8)))  _Float16 v8h;
typedef __attribute__((ext_vector_type(8)))  float    v8f;

#define NB    16
#define LSEQ  512
#define LPAD  514                 // 1 zero halo row each side -> branch-free conv
#define EDIM  256
#define FDIM  256
#define MMEL  4096
#define KDIM  768                 // 3 * 256 (im2col K)
#define MROWS (NB*LSEQ)           // 8192

// ------------------------------------------------------------------
// f32 -> f16 convert into padded [NB][LPAD][EDIM] layout, zeroing halos.
// ------------------------------------------------------------------
__global__ void cvt_pad_kernel(const float* __restrict__ src,
                               _Float16* __restrict__ dst)
{
    int idx = blockIdx.x * blockDim.x + threadIdx.x;      // over NB*LPAD*EDIM
    if (idx >= NB * LPAD * EDIM) return;
    int e = idx & 255;
    int r = (idx >> 8) % LPAD;
    int n = (idx >> 8) / LPAD;
    float v = 0.f;
    if (r >= 1 && r <= LSEQ)
        v = src[((size_t)(n * LSEQ + r - 1) << 8) + e];
    dst[idx] = (_Float16)v;
}

// Zero the halo rows of a padded activation buffer (for h1 before layer 2).
__global__ void zero_pad_kernel(_Float16* __restrict__ dst)
{
    int idx = blockIdx.x * blockDim.x + threadIdx.x;      // NB*2*EDIM = 8192
    if (idx >= NB * 2 * EDIM) return;
    int e = idx & 255;
    int k = idx >> 8;                                     // 0..31
    int n = k >> 1;
    int r = (k & 1) ? (LPAD - 1) : 0;
    dst[((size_t)n * LPAD + r) * EDIM + e] = (_Float16)0.f;
}

// ------------------------------------------------------------------
// Pack conv weight [F][E][3] (f32) into WMMA B-fragment layout (f16).
// im2col K = shift*256 + e.  Fragment: col = lane%16, h = lane/16;
// elem i<8 -> K = 8h+i, elem i>=8 -> K = 16+8h+(i-8).
// flat index = ((kc*16 + nt)*32 + lane)*16 + i
// ------------------------------------------------------------------
__global__ void pack_w_kernel(const float* __restrict__ w,
                              _Float16* __restrict__ pack)
{
    int idx = blockIdx.x * blockDim.x + threadIdx.x;
    if (idx >= (KDIM/32) * (FDIM/16) * 32 * 16) return;
    int i    =  idx        & 15;
    int lane = (idx >> 4)  & 31;
    int nt   = (idx >> 9)  & 15;
    int kc   =  idx >> 13;
    int klocal = ((lane >> 4) << 3) + (i & 7) + ((i >> 3) << 4);
    int kglob  = kc * 32 + klocal;
    int shift  = kglob >> 8;
    int e      = kglob & 255;
    int f      = nt * 16 + (lane & 15);
    pack[idx] = (_Float16)w[(f * EDIM + e) * 3 + shift];
}

// ------------------------------------------------------------------
// Fused conv1d(k=3,same) GEMM + bias + LayerNorm + ReLU.
// Block: 256 thr (8 waves) -> 64 rows x 256 cols.  Wave: 32 rows x 64 cols
// (2 M-tiles x 4 N-tiles).  Register double-buffered K loop (24 chunks of 32),
// pipeline pinned with sched_barrier so prefetches cannot sink to use sites.
// LN stats from accumulators via intra-16-lane shuffles + tiny LDS partials.
// LAYER==1 writes f16 activations (padded); LAYER==2 fuses lin_w dot -> dur.
// ------------------------------------------------------------------
template<int LAYER>
__global__ __launch_bounds__(256)
void conv_ln_kernel(const _Float16* __restrict__ A,    // padded [NB][514][256] f16
                    const _Float16* __restrict__ Bp,   // packed weights
                    const float* __restrict__ bias,
                    const float* __restrict__ lng,
                    const float* __restrict__ lnb,
                    _Float16* __restrict__ out16,      // LAYER==1 (padded)
                    const float* __restrict__ linw,    // LAYER==2
                    const float* __restrict__ linb,    // LAYER==2
                    float* __restrict__ dur)           // LAYER==2
{
    __shared__ float psum [64][4];
    __shared__ float psum2[64][4];
    __shared__ float rstat[64][2];
    __shared__ float pdot [64][4];

    const int lane = threadIdx.x & 31;
    const int wv   = threadIdx.x >> 5;    // 0..7
    const int wrow = wv & 1;              // which 32-row half
    const int wcol = wv >> 1;             // which 64-col slice
    const int m0   = blockIdx.x * 64;     // 128 blocks; block stays in one seq
    const int seq  = m0 >> 9;
    const int l0   = (m0 & 511) + wrow * 32 + (lane & 15);
    const int eoff = (lane >> 4) << 3;    // A K-chunk select per lane half

    // Padded A base at halo row 0: row for (l, shift) = l + shift.
    const _Float16* abase = A + (size_t)seq * LPAD * EDIM + (size_t)l0 * EDIM + eoff;
    const _Float16* bbase = Bp + (size_t)(wcol * 4 * 32 + lane) * 16;

    v8f c[2][4] = {};

    union AF { v16h v; v8h h[2]; };
    AF   a[2][2];      // [buf][mtile]
    v16h b[2][4];      // [buf][ntile]

    auto lda = [&](AF* dst, int kc) {
        const _Float16* rp = abase + (kc >> 3) * EDIM + (kc & 7) * 32;
        dst[0].h[0] = *(const v8h*)(rp);
        dst[0].h[1] = *(const v8h*)(rp + 16);
        dst[1].h[0] = *(const v8h*)(rp + 16 * EDIM);
        dst[1].h[1] = *(const v8h*)(rp + 16 * EDIM + 16);
    };
    auto ldb = [&](v16h* dst, int kc) {
        const _Float16* bp = bbase + (size_t)kc * 8192;   // kc*16*32*16 halves
        dst[0] = *(const v16h*)(bp);
        dst[1] = *(const v16h*)(bp + 512);
        dst[2] = *(const v16h*)(bp + 1024);
        dst[3] = *(const v16h*)(bp + 1536);
    };

    lda(a[0], 0);
    ldb(b[0], 0);
    #pragma unroll
    for (int kc = 0; kc < KDIM / 32; ++kc) {
        const int cur = kc & 1, nxt = cur ^ 1;
        if (kc + 1 < KDIM / 32) {            // prefetch next chunk first
            lda(a[nxt], kc + 1);
            ldb(b[nxt], kc + 1);
        }
        // Pin pipeline: prefetches for chunk kc+1 must issue before the
        // WMMAs of chunk kc; WMMAs of chunk kc stay between this barrier
        // and the next one (may interleave with chunk kc+2 loads).
        __builtin_amdgcn_sched_barrier(0);
        #pragma unroll
        for (int t = 0; t < 4; ++t) {
            c[0][t] = __builtin_amdgcn_wmma_f32_16x16x32_f16(
                          false, a[cur][0].v, false, b[cur][t], (short)0,
                          c[0][t], false, false);
            c[1][t] = __builtin_amdgcn_wmma_f32_16x16x32_f16(
                          false, a[cur][1].v, false, b[cur][t], (short)0,
                          c[1][t], false, false);
        }
    }

    // ---- LayerNorm stats straight from accumulators ----
    // C layout: elem i -> row (lane>=16 ? 8:0)+i (+16*mt), col = lane%16 (+16*t)
    const int colb = wcol * 64 + (lane & 15);
    float bias_t[4];
    #pragma unroll
    for (int t = 0; t < 4; ++t) bias_t[t] = bias[colb + t * 16];

    const int rgrp = wrow * 32 + ((lane >> 4) << 3);
    #pragma unroll
    for (int mt = 0; mt < 2; ++mt) {
        #pragma unroll
        for (int i = 0; i < 8; ++i) {
            float s = 0.f, s2 = 0.f;
            #pragma unroll
            for (int t = 0; t < 4; ++t) {
                float v = c[mt][t][i] + bias_t[t];
                s += v; s2 += v * v;
            }
            #pragma unroll
            for (int m = 8; m > 0; m >>= 1) {   // reduce within 16-lane group
                s  += __shfl_xor(s,  m, 32);
                s2 += __shfl_xor(s2, m, 32);
            }
            if ((lane & 15) == 0) {
                int row = rgrp + mt * 16 + i;
                psum [row][wcol] = s;
                psum2[row][wcol] = s2;
            }
        }
    }
    __syncthreads();
    if (threadIdx.x < 64) {
        float s  = psum [threadIdx.x][0] + psum [threadIdx.x][1]
                 + psum [threadIdx.x][2] + psum [threadIdx.x][3];
        float s2 = psum2[threadIdx.x][0] + psum2[threadIdx.x][1]
                 + psum2[threadIdx.x][2] + psum2[threadIdx.x][3];
        float mean = s * (1.f / 256.f);
        float var  = s2 * (1.f / 256.f) - mean * mean;
        rstat[threadIdx.x][0] = mean;
        rstat[threadIdx.x][1] = rsqrtf(var + 1e-5f);
    }
    __syncthreads();

    // ---- normalize + ReLU (+ store f16 / + lin_w dot) ----
    float g_t[4], bb_t[4], lw_t[4];
    #pragma unroll
    for (int t = 0; t < 4; ++t) {
        g_t[t]  = lng[colb + t * 16];
        bb_t[t] = lnb[colb + t * 16];
        lw_t[t] = (LAYER == 2) ? linw[colb + t * 16] : 0.f;
    }
    #pragma unroll
    for (int mt = 0; mt < 2; ++mt) {
        #pragma unroll
        for (int i = 0; i < 8; ++i) {
            const int row  = rgrp + mt * 16 + i;       // block-local 0..63
            const float mean = rstat[row][0];
            const float rs   = rstat[row][1];
            if (LAYER == 1) {
                const int gl = (m0 + row) & 511;
                _Float16* op = out16 + (size_t)seq * LPAD * EDIM
                                     + (size_t)(gl + 1) * EDIM + colb;
                #pragma unroll
                for (int t = 0; t < 4; ++t) {
                    float hv = (c[mt][t][i] + bias_t[t] - mean) * rs * g_t[t] + bb_t[t];
                    op[t * 16] = (_Float16)fmaxf(hv, 0.f);
                }
            } else {
                float d = 0.f;
                #pragma unroll
                for (int t = 0; t < 4; ++t) {
                    float hv = (c[mt][t][i] + bias_t[t] - mean) * rs * g_t[t] + bb_t[t];
                    d += fmaxf(hv, 0.f) * lw_t[t];
                }
                #pragma unroll
                for (int m = 8; m > 0; m >>= 1) d += __shfl_xor(d, m, 32);
                if ((lane & 15) == 0) pdot[row][wcol] = d;
            }
        }
    }
    if (LAYER == 2) {
        __syncthreads();
        if (threadIdx.x < 64) {
            float d = pdot[threadIdx.x][0] + pdot[threadIdx.x][1]
                    + pdot[threadIdx.x][2] + pdot[threadIdx.x][3];
            dur[m0 + threadIdx.x] = fmaxf(d + linb[0], 0.f);
        }
    }
}

// ------------------------------------------------------------------
// Length regulator: exclusive cumsum of durations (16 tiny serial scans)
// ------------------------------------------------------------------
__global__ void cumsum_kernel(const int* __restrict__ tgt, int* __restrict__ start)
{
    int n = threadIdx.x;
    if (n >= NB) return;
    int c = 0;
    for (int l = 0; l < LSEQ; ++l) {
        int t = tgt[n * LSEQ + l];
        start[n * LSEQ + l] = c;
        c += t;
    }
}

__global__ void zero_kernel(float4* __restrict__ p, int n4)
{
    int i = blockIdx.x * blockDim.x + threadIdx.x;
    if (i < n4) { float4 z = {0.f, 0.f, 0.f, 0.f}; p[i] = z; }
}

// One block per (n,l): copy x[n,l,:] (fp32, exact as reference) into
// out rows [start, start+dur). Intervals disjoint -> no collisions.
__global__ void scatter_kernel(const float* __restrict__ x,
                               const int* __restrict__ tgt,
                               const int* __restrict__ start,
                               float* __restrict__ out)
{
    const int m = blockIdx.x;             // n*512 + l
    const int n = m >> 9;
    const float4 v = ((const float4*)(x + (size_t)m * EDIM))[threadIdx.x];
    const int s = start[m];
    const int d = tgt[m];
    float4* ob = (float4*)(out + (size_t)n * MMEL * EDIM);
    for (int r = 0; r < d; ++r)
        ob[(size_t)(s + r) * (EDIM / 4) + threadIdx.x] = v;
}

// ------------------------------------------------------------------
extern "C" void kernel_launch(void* const* d_in, const int* in_sizes, int n_in,
                              void* d_out, int out_size, void* d_ws, size_t ws_size,
                              hipStream_t stream)
{
    const float* x       = (const float*)d_in[0];
    const float* conv1_w = (const float*)d_in[1];
    const float* conv1_b = (const float*)d_in[2];
    const float* ln1_g   = (const float*)d_in[3];
    const float* ln1_b   = (const float*)d_in[4];
    const float* conv2_w = (const float*)d_in[5];
    const float* conv2_b = (const float*)d_in[6];
    const float* ln2_g   = (const float*)d_in[7];
    const float* ln2_b   = (const float*)d_in[8];
    const float* lin_w   = (const float*)d_in[9];
    const float* lin_b   = (const float*)d_in[10];
    const int*   target  = (const int*)d_in[11];
    // d_in[12] = mel_max_length (4096, baked into MMEL)

    const size_t XPAD_BYTES = (size_t)NB * LPAD * EDIM * sizeof(_Float16); // 4,210,688
    const size_t WPK_BYTES  = (size_t)KDIM * FDIM * sizeof(_Float16);      // 393,216

    char* ws = (char*)d_ws;
    _Float16* x16  = (_Float16*)(ws);
    _Float16* h116 = (_Float16*)(ws + XPAD_BYTES);
    _Float16* w1p  = (_Float16*)(ws + 2 * XPAD_BYTES);
    _Float16* w2p  = (_Float16*)(ws + 2 * XPAD_BYTES + WPK_BYTES);
    int*      strt = (int*)     (ws + 2 * XPAD_BYTES + 2 * WPK_BYTES);

    float* out_main = (float*)d_out;                        // [16,4096,256]
    float* dur      = out_main + (size_t)NB * MMEL * EDIM;  // [16,512]

    // ---- duration predictor (WMMA path) ----
    const int xpcnt = NB * LPAD * EDIM;
    cvt_pad_kernel<<<(xpcnt + 255) / 256, 256, 0, stream>>>(x, x16);
    pack_w_kernel<<<(KDIM * FDIM + 255) / 256, 256, 0, stream>>>(conv1_w, w1p);
    pack_w_kernel<<<(KDIM * FDIM + 255) / 256, 256, 0, stream>>>(conv2_w, w2p);
    zero_pad_kernel<<<(NB * 2 * EDIM + 255) / 256, 256, 0, stream>>>(h116);
    conv_ln_kernel<1><<<MROWS / 64, 256, 0, stream>>>(
        x16, w1p, conv1_b, ln1_g, ln1_b, h116, nullptr, nullptr, nullptr);
    conv_ln_kernel<2><<<MROWS / 64, 256, 0, stream>>>(
        h116, w2p, conv2_b, ln2_g, ln2_b, nullptr, lin_w, lin_b, dur);

    // ---- length regulation (bandwidth path) ----
    cumsum_kernel<<<1, 32, 0, stream>>>(target, strt);
    const int n4 = NB * MMEL * EDIM / 4;
    zero_kernel<<<(n4 + 255) / 256, 256, 0, stream>>>((float4*)out_main, n4);
    scatter_kernel<<<NB * LSEQ, EDIM / 4, 0, stream>>>(x, target, strt, out_main);
}
